// MultiHeadedAttention_50534585204965
// MI455X (gfx1250) — matrix operand order repaired and checked by reading, not compiled, not run on hardware
//
#include <hip/hip_runtime.h>

// Problem constants (from reference): B=2, T=2048, E=1024, H=16, HD=64
#define Bc   2
#define Tc   2048
#define Ec   1024
#define Hc   16
#define HDc  64
#define BTc  (Bc * Tc)          // 4096 rows of X
#define BHc  (Bc * Hc)          // 32 (b,h) pairs
#define BHTD ((size_t)BHc * Tc * HDc)   // 4,194,304 bf16 elems per Q/K/V buffer

typedef __bf16 bf16;
typedef __attribute__((ext_vector_type(16))) __bf16 bf16x16;
typedef __attribute__((ext_vector_type(8)))  __bf16 bf16x8;
typedef __attribute__((ext_vector_type(4)))  __bf16 bf16x4;
typedef __attribute__((ext_vector_type(8)))  float  f32x8;
typedef __attribute__((ext_vector_type(4)))  float  f32x4;

// D = A(16x32 bf16) * B(32x16 bf16) + C(16x16 f32), wave32 WMMA
__device__ __forceinline__ f32x8 wmma_bf16(bf16x16 a, bf16x16 b, f32x8 c) {
  return __builtin_amdgcn_wmma_f32_16x16x32_bf16(false, a, false, b, (short)0, c,
                                                 false, false);
}

// Async global->LDS DMA (CDNA5 GLOBAL_LOAD_ASYNC_TO_LDS_B128, ASYNCcnt).
// LDS byte address = low 32 bits of the generic pointer (addrspace(3) offset).
__device__ __forceinline__ void async_copy_b128(void* ldsDst, const void* gSrc) {
  const unsigned l = (unsigned)(unsigned long long)ldsDst;
  asm volatile("global_load_async_to_lds_b128 %0, %1, off"
               :
               : "v"(l), "v"(gSrc)
               : "memory");
}
__device__ __forceinline__ void wait_async0() {
  asm volatile("s_wait_asynccnt 0" ::: "memory");
}

// A-matrix fragment: lane row = lane&15; elems 0..7 at k = kOff+8*half,
// elems 8..15 at k = kOff+16+8*half (ISA 7.12.2, 16-bit A 16x32).
template <int S>
__device__ __forceinline__ bf16x16 frag_a(bf16 M[][S], int row, int kOff, int half) {
  union { bf16x16 v; bf16x8 h[2]; } u;
  u.h[0] = *(const bf16x8*)&M[row][kOff + 8 * half];
  u.h[1] = *(const bf16x8*)&M[row][kOff + 16 + 8 * half];
  return u.v;
}
// B-matrix fragment: lane col = lane&15 (stored as LDS row); elems 0..7 at
// k = kOff+8*half, elems 8..15 at k = kOff+16+8*half (same order as A).
template <int S>
__device__ __forceinline__ bf16x16 frag_b(bf16 M[][S], int row, int kOff, int half) {
  union { bf16x16 v; bf16x8 h[2]; } u;
  u.h[0] = *(const bf16x8*)&M[row][kOff + 8 * half];
  u.h[1] = *(const bf16x8*)&M[row][kOff + 16 + 8 * half];
  return u.v;
}

// ---------------------------------------------------------------------------
// Kernel 1: Y = X @ W^T + b for W in {Wq,Wk,Wv}; store bf16 as [B,H,T,HD].
// Block = 256 threads (8 waves, 2x4), tile 128(M) x 128(N), K-step 32.
// f32->bf16 conversion happens in flight, so this one keeps the
// global_load_b128 -> v_cvt_pk_bf16_f32 -> ds_store staging path.
// ---------------------------------------------------------------------------
__global__ __launch_bounds__(256) void qkv_proj_kernel(
    const float* __restrict__ X,
    const float* __restrict__ Wq, const float* __restrict__ bq,
    const float* __restrict__ Wk, const float* __restrict__ bk,
    const float* __restrict__ Wv, const float* __restrict__ bv,
    bf16* __restrict__ Qb, bf16* __restrict__ Kb, bf16* __restrict__ Vb) {
  __shared__ bf16 As[128][40];   // X tile   (rows m, k)   80B row stride
  __shared__ bf16 Bs[128][40];   // W tile   (rows n, k)   (B = W^T)

  const int tid  = threadIdx.x;
  const int lane = tid & 31;
  const int wid  = tid >> 5;
  const int half = lane >> 4;
  const int r    = lane & 15;
  const int m0   = blockIdx.x * 128;
  const int n0   = blockIdx.y * 128;

  const float* W;  const float* bias;  bf16* dst;
  if (blockIdx.z == 0)      { W = Wq; bias = bq; dst = Qb; }
  else if (blockIdx.z == 1) { W = Wk; bias = bk; dst = Kb; }
  else                      { W = Wv; bias = bv; dst = Vb; }

  const int wm = wid & 1;   // 0..1 -> M offset wm*64
  const int wn = wid >> 1;  // 0..3 -> N offset wn*32

  f32x8 acc[4][2] = {};     // 4 M-tiles x 2 N-tiles of 16x16

  const int srow = tid >> 3;          // 0..31
  const int sc4  = (tid & 7) * 4;     // 0..28

  for (int k0 = 0; k0 < Ec; k0 += 32) {
    __syncthreads();
#pragma unroll
    for (int it = 0; it < 4; ++it) {
      const int rr = srow + it * 32;
      f32x4 xa = *(const f32x4*)&X[(size_t)(m0 + rr) * Ec + k0 + sc4];
      f32x4 wa = *(const f32x4*)&W[(size_t)(n0 + rr) * Ec + k0 + sc4];
      bf16x4 xb, wb;
#pragma unroll
      for (int j = 0; j < 4; ++j) { xb[j] = (bf16)xa[j]; wb[j] = (bf16)wa[j]; }
      *(bf16x4*)&As[rr][sc4] = xb;
      *(bf16x4*)&Bs[rr][sc4] = wb;
      if (k0 + 32 < Ec) {  // pull next K-slice toward L2 (global_prefetch_b8)
        __builtin_prefetch(&X[(size_t)(m0 + rr) * Ec + k0 + 32 + sc4], 0, 1);
        __builtin_prefetch(&W[(size_t)(n0 + rr) * Ec + k0 + 32 + sc4], 0, 1);
      }
    }
    __syncthreads();

    bf16x16 af[4], bfrag[2];
#pragma unroll
    for (int mt = 0; mt < 4; ++mt) af[mt]    = frag_a(As, wm * 64 + mt * 16 + r, 0, half);
#pragma unroll
    for (int nt = 0; nt < 2; ++nt) bfrag[nt] = frag_b(Bs, wn * 32 + nt * 16 + r, 0, half);
#pragma unroll
    for (int mt = 0; mt < 4; ++mt)
#pragma unroll
      for (int nt = 0; nt < 2; ++nt)
        acc[mt][nt] = wmma_bf16(af[mt], bfrag[nt], acc[mt][nt]);
  }

  // bias + store bf16 into [B,H,T,HD]
#pragma unroll
  for (int nt = 0; nt < 2; ++nt) {
    const int   ge = n0 + wn * 32 + nt * 16 + r;   // column in E
    const float bb = bias[ge];
    const int   h  = ge >> 6, d = ge & 63;
#pragma unroll
    for (int mt = 0; mt < 4; ++mt) {
      const int gmBase = m0 + wm * 64 + mt * 16 + 8 * half;
#pragma unroll
      for (int i = 0; i < 8; ++i) {
        const int gm = gmBase + i;
        const int b  = gm >> 11, t = gm & (Tc - 1);
        dst[(((size_t)(b * Hc + h)) * Tc + t) * HDc + d] =
            (bf16)(acc[mt][nt][i] + bb);
      }
    }
  }
}

// ---------------------------------------------------------------------------
// Kernel 2: per-key-COLUMN softmax stats (softmax is over the query axis).
// Block owns 128 key columns of one (b,h); loops q-blocks of 128 from kBase.
// Online (m,l) per column; D-layout column reduce = 8 VGPRs + shfl_xor(16).
// Q/K tiles are already bf16 -> staged with async global->LDS DMA.
// ---------------------------------------------------------------------------
__global__ __launch_bounds__(256) void colstats_kernel(
    const bf16* __restrict__ Qb, const bf16* __restrict__ Kb,
    float* __restrict__ colM, float* __restrict__ colL) {
  __shared__ bf16  Ks[128][72];   // K rows for the 128 columns (144B stride)
  __shared__ bf16  Qs[128][72];
  __shared__ float sM[8][128];
  __shared__ float sL[8][128];

  const int tid  = threadIdx.x;
  const int lane = tid & 31;
  const int wid  = tid >> 5;
  const int half = lane >> 4;
  const int r    = lane & 15;
  const int kBase = blockIdx.x * 128;
  const int bh    = blockIdx.y;
  const size_t base = (size_t)bh * Tc * HDc;

  {  // stage the 128 key vectors once (async DMA, ASYNCcnt-tracked)
    const int row = tid >> 1, seg = tid & 1;
#pragma unroll
    for (int c = 0; c < 4; ++c)
      async_copy_b128(&Ks[row][seg * 32 + c * 8],
                      &Kb[base + (size_t)(kBase + row) * HDc + seg * 32 + c * 8]);
  }

  float mRun[8], lRun[8];
#pragma unroll
  for (int nt = 0; nt < 8; ++nt) { mRun[nt] = -3.0e38f; lRun[nt] = 0.f; }
  const float scale = 0.03125f;  // 1/sqrt(E) = 1/32

  for (int qb = kBase; qb < Tc; qb += 128) {
    __syncthreads();
    {  // stage 128 query rows (async DMA)
      const int row = tid >> 1, seg = tid & 1;
#pragma unroll
      for (int c = 0; c < 4; ++c)
        async_copy_b128(&Qs[row][seg * 32 + c * 8],
                        &Qb[base + (size_t)(qb + row) * HDc + seg * 32 + c * 8]);
    }
    wait_async0();       // our LDS writes have landed (also covers Ks 1st iter)
    __syncthreads();     // everyone's writes visible

    bf16x16 a0 = frag_a(Qs, wid * 16 + r, 0, half);
    bf16x16 a1 = frag_a(Qs, wid * 16 + r, 32, half);
    const bool maskBlk = (qb == kBase);  // only the diagonal q-block masks
    const int  qBase   = qb + wid * 16 + 8 * half;

#pragma unroll
    for (int nt = 0; nt < 8; ++nt) {
      f32x8 acc = {};
      acc = wmma_bf16(a0, frag_b(Ks, nt * 16 + r, 0, half), acc);
      acc = wmma_bf16(a1, frag_b(Ks, nt * 16 + r, 32, half), acc);

      const int kIdx = kBase + nt * 16 + r;   // this lane's column
      float s[8];  bool val[8];  float tm = -3.0e38f;
#pragma unroll
      for (int i = 0; i < 8; ++i) {
        val[i] = !maskBlk || (kIdx <= qBase + i);
        s[i]   = val[i] ? acc[i] * scale : -3.0e38f;
        tm     = fmaxf(tm, s[i]);
      }
      tm = fmaxf(tm, __shfl_xor(tm, 16, 32));      // combine the two m-halves
      const float mNew = fmaxf(mRun[nt], tm);
      float ps = 0.f;
#pragma unroll
      for (int i = 0; i < 8; ++i) ps += val[i] ? __expf(s[i] - mNew) : 0.f;
      ps += __shfl_xor(ps, 16, 32);
      lRun[nt] = lRun[nt] * __expf(mRun[nt] - mNew) + ps;
      mRun[nt] = mNew;
    }
  }

  if (half == 0) {
#pragma unroll
    for (int nt = 0; nt < 8; ++nt) {
      sM[wid][nt * 16 + r] = mRun[nt];
      sL[wid][nt * 16 + r] = lRun[nt];
    }
  }
  __syncthreads();
  if (tid < 128) {
    float m = -3.0e38f;
#pragma unroll
    for (int w = 0; w < 8; ++w) m = fmaxf(m, sM[w][tid]);
    float l = 0.f;
#pragma unroll
    for (int w = 0; w < 8; ++w) l += sL[w][tid] * __expf(sM[w][tid] - m);
    colM[(size_t)bh * Tc + kBase + tid] = m;
    colL[(size_t)bh * Tc + kBase + tid] = l;
  }
}

// ---------------------------------------------------------------------------
// Kernel 3: out[q,:] = sum_{k<=q} exp(s[q,k]-m_k)/l_k * V[k,:].
// Block owns 128 q rows of one (b,h); k-loop in steps of 32; P goes through
// LDS (wave-local rows) to convert D-layout -> A-layout for the PV GEMM.
// Q/K staged via async DMA; V needs an in-flight transpose so it stays on
// the load->ds_store path.
// ---------------------------------------------------------------------------
__global__ __launch_bounds__(256) void attn_out_kernel(
    const bf16* __restrict__ Qb, const bf16* __restrict__ Kb,
    const bf16* __restrict__ Vb,
    const float* __restrict__ colM, const float* __restrict__ colL,
    float* __restrict__ out) {
  __shared__ bf16 Qs[128][72];
  __shared__ bf16 Ks[32][72];
  __shared__ bf16 Vs[64][40];    // V transposed: rows d, cols k
  __shared__ bf16 Ps[128][40];   // P tile: rows q, cols k (wave-local rows)

  const int tid  = threadIdx.x;
  const int lane = tid & 31;
  const int wid  = tid >> 5;
  const int half = lane >> 4;
  const int r    = lane & 15;
  const int q0 = blockIdx.x * 128;
  const int bh = blockIdx.y;
  const int b  = bh >> 4, h = bh & 15;
  const size_t base = (size_t)bh * Tc * HDc;

  {  // stage the 128 query rows once (async DMA)
    const int row = tid >> 1, seg = tid & 1;
#pragma unroll
    for (int c = 0; c < 4; ++c)
      async_copy_b128(&Qs[row][seg * 32 + c * 8],
                      &Qb[base + (size_t)(q0 + row) * HDc + seg * 32 + c * 8]);
  }

  f32x8 oAcc[4] = {};
  const float scale = 0.03125f;
  const int kEnd = q0 + 128;                 // causal: k <= q0+127
  const int qBase = q0 + wid * 16 + 8 * half;

  for (int k0 = 0; k0 < kEnd; k0 += 32) {
    __syncthreads();
    {  // stage K rows (async DMA) and V transposed for this k-slab
      const int row = tid >> 3, seg = tid & 7;   // row 0..31, seg 0..7
      async_copy_b128(&Ks[row][seg * 8],
                      &Kb[base + (size_t)(k0 + row) * HDc + seg * 8]);
      bf16x8 vv = *(const bf16x8*)&Vb[base + (size_t)(k0 + row) * HDc + seg * 8];
#pragma unroll
      for (int j = 0; j < 8; ++j) Vs[seg * 8 + j][row] = vv[j];
    }
    wait_async0();       // Ks (and first-iter Qs) landed in LDS
    __syncthreads();

    bf16x16 aq0 = frag_a(Qs, wid * 16 + r, 0, half);
    bf16x16 aq1 = frag_a(Qs, wid * 16 + r, 32, half);

#pragma unroll
    for (int nt = 0; nt < 2; ++nt) {
      f32x8 acc = {};
      acc = wmma_bf16(aq0, frag_b(Ks, nt * 16 + r, 0, half), acc);
      acc = wmma_bf16(aq1, frag_b(Ks, nt * 16 + r, 32, half), acc);

      const int   kIdx = k0 + nt * 16 + r;
      const float m    = colM[(size_t)bh * Tc + kIdx];
      const float rl   = 1.0f / colL[(size_t)bh * Tc + kIdx];
#pragma unroll
      for (int i = 0; i < 8; ++i) {
        const bool  valid = (kIdx <= qBase + i);
        const float p = valid ? __expf(acc[i] * scale - m) * rl : 0.f;
        Ps[wid * 16 + 8 * half + i][nt * 16 + r] = (bf16)p;
      }
    }
    // wave-local LDS RAW (wrote our own 16 q-rows of Ps); DS is in-order per
    // wave — make it explicit and fence the compiler.
    asm volatile("s_wait_dscnt 0" ::: "memory");

    bf16x16 ap = frag_a(Ps, wid * 16 + r, 0, half);
#pragma unroll
    for (int dt = 0; dt < 4; ++dt)
      oAcc[dt] = wmma_bf16(ap, frag_b(Vs, dt * 16 + r, 0, half), oAcc[dt]);
  }

  // store fp32 output [B,T,E]
#pragma unroll
  for (int dt = 0; dt < 4; ++dt) {
    const int d = dt * 16 + r;
#pragma unroll
    for (int i = 0; i < 8; ++i) {
      const int t = q0 + wid * 16 + 8 * half + i;
      out[((size_t)(b * Tc + t)) * Ec + h * HDc + d] = oAcc[dt][i];
    }
  }
}

// ---------------------------------------------------------------------------
extern "C" void kernel_launch(void* const* d_in, const int* in_sizes, int n_in,
                              void* d_out, int out_size, void* d_ws, size_t ws_size,
                              hipStream_t stream) {
  (void)in_sizes; (void)n_in; (void)out_size; (void)ws_size;
  const float* X  = (const float*)d_in[0];
  const float* Wq = (const float*)d_in[1];
  const float* bq = (const float*)d_in[2];
  const float* Wk = (const float*)d_in[3];
  const float* bk = (const float*)d_in[4];
  const float* Wv = (const float*)d_in[5];
  const float* bv = (const float*)d_in[6];
  float* out = (float*)d_out;

  bf16*  Qb   = (bf16*)d_ws;
  bf16*  Kb   = Qb + BHTD;
  bf16*  Vb   = Kb + BHTD;
  float* colM = (float*)(Vb + BHTD);
  float* colL = colM + (size_t)BHc * Tc;

  qkv_proj_kernel<<<dim3(BTc / 128, Ec / 128, 3), 256, 0, stream>>>(
      X, Wq, bq, Wk, bk, Wv, bv, Qb, Kb, Vb);
  colstats_kernel<<<dim3(Tc / 128, BHc), 256, 0, stream>>>(Qb, Kb, colM, colL);
  attn_out_kernel<<<dim3(Tc / 128, BHc), 256, 0, stream>>>(Qb, Kb, Vb, colM, colL, out);
}
